// TargetRegion_av2_multiagent_8280696947151
// MI455X (gfx1250) — compile-verified, matrix operand order checked by reading, not run on hardware
//
#include <hip/hip_runtime.h>
#include <hip/hip_bf16.h>
#include <math.h>

// ---------------- problem constants ----------------
#define S_  128
#define V_  64
#define M_  6
#define L_  1000
#define N_  (S_ * V_)        // 8192
#define NM_ (N_ * M_)        // 49152
#define H_  64
#define SMVV ((size_t)S_ * M_ * V_ * V_)   // 3,145,728

// output offsets (floats), tuple order:
// score, dist_all, offset_all, th, lane_embed, node_tar_index, hardmask
#define OFF_SCORE ((size_t)0)
#define OFF_DIST  ((size_t)49152)
#define OFF_OFFS  (OFF_DIST + SMVV * 4)            // 12,632,064
#define OFF_TH    (OFF_OFFS + SMVV * 4 * 2)        // 37,797,888
#define OFF_LANE  (OFF_TH + (size_t)NM_ * 4 * 2)   // 38,191,104
#define OFF_IDX   (OFF_LANE + (size_t)S_ * L_ * 64)// 46,383,104
#define OFF_MASK  (OFF_IDX + SMVV * 2)             // 52,674,560

typedef __attribute__((ext_vector_type(16))) _Float16 v16h;
typedef __attribute__((ext_vector_type(8)))  float    v8f;

// K index held by fragment element e for lane-half `half` (ISA 7.12.2, 16-bit A/B)
__device__ __forceinline__ int frag_k(int e, int half) {
  return e + 8 * half + ((e & 8) ? 8 : 0);
}

// A (16x32 f16) fragment from row-major LDS tile: two contiguous 16B runs per lane
__device__ __forceinline__ v16h lds_load_a(const _Float16* A, int lda, int row0, int k0) {
  const int lane = threadIdx.x & 31;
  const int m = lane & 15, half = lane >> 4;
  const _Float16* p = A + (size_t)(row0 + m) * lda + k0 + 8 * half;
  v16h a;
#pragma unroll
  for (int e = 0; e < 8; ++e)  a[e] = p[e];
#pragma unroll
  for (int e = 8; e < 16; ++e) a[e] = p[e + 8];
  return a;
}

// Block GEMM: 128 rows (8 waves x 16), A in LDS (f16, 128 x KPAD, zero-padded).
// Weights KxN f32 in global; each 16-col panel is staged into LDS PRE-SWIZZLED
// into the B-fragment lane layout, so each lane reads its fragment as one
// aligned 32-byte block (2x ds_load_b128). A fragments are hoisted across the
// n-tile loop. Bias + optional relu; writes f16 LDS tile and/or f32 global.
template <int KPAD, int NPAD>
__device__ __forceinline__ void gemm128(const _Float16* __restrict__ A,
                                        const float* __restrict__ Wg, int K, int N,
                                        _Float16* __restrict__ wpanel,  // LDS, KSTEPS*512 f16
                                        const float* __restrict__ bias, bool relu,
                                        _Float16* __restrict__ outL,
                                        float* __restrict__ outG, int ldg) {
  constexpr int KSTEPS = KPAD / 32;
  constexpr int NTILES = NPAD / 16;
  const int wave = threadIdx.x >> 5;
  const int lane = threadIdx.x & 31;
  const int row0 = wave << 4;

  // hoist A fragments (invariant across n-tiles)
  v16h afrag[KSTEPS];
#pragma unroll
  for (int ks = 0; ks < KSTEPS; ++ks)
    afrag[ks] = lds_load_a(A, KPAD, row0, ks << 5);

  for (int nt = 0; nt < NTILES; ++nt) {
    __syncthreads();                       // previous panel fully consumed
    const int n0 = nt << 4;
    // stage panel pre-swizzled: wpanel[ks*512 + l*16 + e] = W[ks*32+frag_k(e,l>>4)][n0+(l&15)]
    for (int idx = threadIdx.x; idx < KSTEPS * 512; idx += 256) {
      const int ks = idx >> 9;
      const int l  = (idx >> 4) & 31;
      const int e  = idx & 15;
      const int k  = (ks << 5) + frag_k(e, l >> 4);
      const int n  = n0 + (l & 15);
      float v = (k < K && n < N) ? Wg[(size_t)k * N + n] : 0.0f;
      wpanel[idx] = (_Float16)v;
    }
    __syncthreads();
    v8f acc = {};
#pragma unroll
    for (int ks = 0; ks < KSTEPS; ++ks) {
      const v16h b = *(const v16h*)(wpanel + (ks << 9) + lane * 16);
      acc = __builtin_amdgcn_wmma_f32_16x16x32_f16(
          /*neg_a=*/false, afrag[ks], /*neg_b=*/false, b,
          /*c_mod=*/(short)0, acc, /*reuse_a=*/false, /*reuse_b=*/false);
    }
    const int n = n0 + (lane & 15);
    const int rbase = row0 + ((lane >> 4) << 3);
    const bool valid = (n < N);
    const float bv = valid ? bias[n] : 0.0f;
#pragma unroll
    for (int r = 0; r < 8; ++r) {
      float v = valid ? (acc[r] + bv) : 0.0f;
      if (relu) v = fmaxf(v, 0.0f);
      if (outL) outL[(size_t)(rbase + r) * NPAD + n] = (_Float16)v;
      if (outG && valid) outG[(size_t)(rbase + r) * ldg + n] = v;
    }
  }
  __syncthreads();
}

// ---------------- Kernel 1: world transform + th extraction ----------------
__global__ __launch_bounds__(256) void k_world(const float* __restrict__ y_hat,
                                               const float* __restrict__ rot,
                                               const float* __restrict__ cen,
                                               float* __restrict__ world,
                                               float* __restrict__ th_out) {
  int idx = blockIdx.x * 256 + threadIdx.x;          // over NM_*60
  if (idx >= NM_ * 60) return;
  int t = idx % 60;
  int b = idx / 60;                                  // row in N*M
  int nIdx = b / M_;
  float y0 = y_hat[(size_t)idx * 2 + 0];
  float y1 = y_hat[(size_t)idx * 2 + 1];
  const float* R = rot + (size_t)nIdx * 4;           // [c][d] row-major
  float wx = fmaf(y0, R[0], y1 * R[2]) + cen[(size_t)nIdx * 2 + 0];
  float wy = fmaf(y0, R[1], y1 * R[3]) + cen[(size_t)nIdx * 2 + 1];
  world[(size_t)idx * 2 + 0] = wx;
  world[(size_t)idx * 2 + 1] = wy;
  int tq = -1;
  if (t == 14) tq = 0; else if (t == 29) tq = 1;
  else if (t == 44) tq = 2; else if (t == 59) tq = 3;
  if (tq >= 0) {
    th_out[((size_t)b * 4 + tq) * 2 + 0] = wx;
    th_out[((size_t)b * 4 + tq) * 2 + 1] = wy;
  }
}

// ---------------- Kernel 2: lane MLP (3 -> 128 -> 128 -> 64) ----------------
__global__ __launch_bounds__(256) void k_lane(const float* __restrict__ attr,
                                              const float* W1, const float* b1,
                                              const float* W2, const float* b2,
                                              const float* W3, const float* b3,
                                              float* __restrict__ out) {
  __shared__ _Float16 sA[128 * 128];
  __shared__ _Float16 sB[128 * 128];
  __shared__ _Float16 sW[4 * 512];           // max KSTEPS=4 panels
  const int row0 = blockIdx.x * 128;
  for (int idx = threadIdx.x; idx < 128 * 32; idx += 256) {
    int r = idx >> 5, c = idx & 31;
    float v = (c < 3) ? attr[(size_t)(row0 + r) * 3 + c] : 0.0f;
    sA[idx] = (_Float16)v;
  }
  __builtin_prefetch(W2, 0, 1);
  __syncthreads();
  gemm128<32, 128>(sA, W1, 3, 128, sW, b1, true, sB, nullptr, 0);
  __builtin_prefetch(W3, 0, 1);
  gemm128<128, 128>(sB, W2, 128, 128, sW, b2, true, sA, nullptr, 0);
  gemm128<128, 64>(sA, W3, 128, 64, sW, b3, false, nullptr,
                   out + (size_t)row0 * 64, 64);
}

// ---------------- Kernel 3: ego MLP + GRU + score MLP (fused) ----------------
struct FcArgs {
  const float *ego, *W1, *b1, *W2, *b2, *W3, *b3;
  const float *Wih, *bih, *bhh, *pos;
  const float *sW1, *sb1, *sW2, *sb2, *sW3, *sb3;
  float* score_out;
};

__global__ __launch_bounds__(256) void k_fc_gru_score(FcArgs a) {
  __shared__ _Float16 sA[128 * 192];   // input(160) / L2out(160) / gx(192) / score-mid(128)
  __shared__ _Float16 sB[128 * 288];   // L1out(288) / ego(64) / score_in(64) / sL2out(64)
  __shared__ _Float16 sW[9 * 512];     // max KSTEPS=9 panels
  const int row0 = blockIdx.x * 128;
  // input 128 x 134 -> sA 128x160 (zero-padded K)
  for (int idx = threadIdx.x; idx < 128 * 160; idx += 256) {
    int r = idx / 160, c = idx % 160;
    float v = (c < 134) ? a.ego[(size_t)(row0 + r) * 134 + c] : 0.0f;
    sA[idx] = (_Float16)v;
  }
  __builtin_prefetch(a.W1, 0, 1);
  __syncthreads();
  gemm128<160, 288>(sA, a.W1, 134, 268, sW, a.b1, true, sB, nullptr, 0);
  __builtin_prefetch(a.W2, 0, 1);
  gemm128<288, 160>(sB, a.W2, 268, 134, sW, a.b2, true, sA, nullptr, 0);
  __builtin_prefetch(a.W3, 0, 1);
  gemm128<160, 64>(sA, a.W3, 134, 64, sW, a.b3, false, sB, nullptr, 0);
  __builtin_prefetch(a.Wih, 0, 1);
  gemm128<64, 192>(sB, a.Wih, 64, 192, sW, a.bih, false, sA, nullptr, 0); // gx
  // GRU elementwise + pos embed -> score_in (sB, 128x64)
  for (int idx = threadIdx.x; idx < 128 * 64; idx += 256) {
    int r = idx >> 6, c = idx & 63;
    float g0 = (float)sA[r * 192 + c]        + a.bhh[c];
    float g1 = (float)sA[r * 192 + 64 + c]   + a.bhh[64 + c];
    float g2 = (float)sA[r * 192 + 128 + c];
    float rr = 1.0f / (1.0f + expf(-g0));
    float zz = 1.0f / (1.0f + expf(-g1));
    float ng = tanhf(g2 + rr * a.bhh[128 + c]);
    float h  = (1.0f - zz) * ng;
    sB[r * 64 + c] = (_Float16)(h + a.pos[c]);
  }
  __syncthreads();
  gemm128<64, 128>(sB, a.sW1, 64, 128, sW, a.sb1, true, sA, nullptr, 0);
  gemm128<128, 64>(sA, a.sW2, 128, 64, sW, a.sb2, true, sB, nullptr, 0);
  // score layer 3: 64 -> 1 (VALU dot per row)
  if (threadIdx.x < 128) {
    float s = a.sb3[0];
#pragma unroll 8
    for (int k = 0; k < 64; ++k)
      s = fmaf((float)sB[threadIdx.x * 64 + k], a.sW3[k], s);
    a.score_out[row0 + threadIdx.x] = s;
  }
}

// ---------------- Kernel 4: pairwise dist/offset + index/mask ----------------
__global__ __launch_bounds__(256) void k_pairs(const float* __restrict__ world,
                                               const unsigned char* __restrict__ vm,
                                               float* __restrict__ dist_out,
                                               float* __restrict__ off_out,
                                               float* __restrict__ idx_out,
                                               float* __restrict__ mask_out) {
  __shared__ float ys[V_ * 120];                      // 64 agents x 60 steps x 2
  const int s = blockIdx.x / M_;
  const int m = blockIdx.x % M_;
  for (int i = threadIdx.x; i < V_ * 120; i += 256) {
    int v = i / 120, rest = i % 120;
    ys[i] = world[((size_t)((s * V_ + v) * M_) + m) * 120 + rest];
  }
  __syncthreads();
  for (int p = threadIdx.x; p < V_ * V_; p += 256) {
    const int i = p >> 6, j = p & 63;
    const float* yi = ys + i * 120;
    const float* yj = ys + j * 120;
    const size_t base = (size_t)blockIdx.x * (V_ * V_) + p;
#pragma unroll
    for (int seg = 0; seg < 4; ++seg) {
      float best = 3.4e38f, bx = 0.0f, by = 0.0f;
#pragma unroll
      for (int t = 0; t < 15; ++t) {
        const int tt = seg * 15 + t;
        float dx = yi[tt * 2]     - yj[tt * 2];
        float dy = yi[tt * 2 + 1] - yj[tt * 2 + 1];
        float d2 = fmaf(dx, dx, dy * dy);
        if (d2 < best) { best = d2; bx = dx; by = dy; }
      }
      dist_out[(size_t)seg * SMVV + base] = sqrtf(best);
      off_out[((size_t)seg * SMVV + base) * 2 + 0] = bx;
      off_out[((size_t)seg * SMVV + base) * 2 + 1] = by;
    }
    idx_out[base]        = (float)((s * V_ + i) * M_ + m);
    idx_out[SMVV + base] = (float)((s * V_ + j) * M_ + m);
    mask_out[base] = (vm[s * V_ + i] && (i != j)) ? 1.0f : 0.0f;
  }
}

// ---------------- launch ----------------
extern "C" void kernel_launch(void* const* d_in, const int* in_sizes, int n_in,
                              void* d_out, int out_size, void* d_ws, size_t ws_size,
                              hipStream_t stream) {
  (void)in_sizes; (void)n_in; (void)out_size; (void)ws_size;
  const float* y_hat = (const float*)d_in[0];
  const float* ego   = (const float*)d_in[1];
  const float* rot   = (const float*)d_in[2];
  const float* cen   = (const float*)d_in[3];
  const float* attr  = (const float*)d_in[4];
  const unsigned char* vmask = (const unsigned char*)d_in[5];
  const float* fcW1 = (const float*)d_in[6];  const float* fcb1 = (const float*)d_in[7];
  const float* fcW2 = (const float*)d_in[8];  const float* fcb2 = (const float*)d_in[9];
  const float* fcW3 = (const float*)d_in[10]; const float* fcb3 = (const float*)d_in[11];
  const float* lnW1 = (const float*)d_in[12]; const float* lnb1 = (const float*)d_in[13];
  const float* lnW2 = (const float*)d_in[14]; const float* lnb2 = (const float*)d_in[15];
  const float* lnW3 = (const float*)d_in[16]; const float* lnb3 = (const float*)d_in[17];
  const float* scW1 = (const float*)d_in[18]; const float* scb1 = (const float*)d_in[19];
  const float* scW2 = (const float*)d_in[20]; const float* scb2 = (const float*)d_in[21];
  const float* scW3 = (const float*)d_in[22]; const float* scb3 = (const float*)d_in[23];
  const float* Wih  = (const float*)d_in[24];
  /* gru_Whh (d_in[25]) unused: h0 == 0 */
  const float* bih  = (const float*)d_in[26]; const float* bhh = (const float*)d_in[27];
  const float* pos  = (const float*)d_in[28];

  float* out   = (float*)d_out;
  float* world = (float*)d_ws;   // NM_*60*2 f32 scratch (23.6 MB)

  // 1) world transform + th
  k_world<<<(NM_ * 60 + 255) / 256, 256, 0, stream>>>(y_hat, rot, cen, world, out + OFF_TH);

  // 2) lane MLP -> lane_embed
  k_lane<<<(S_ * L_) / 128, 256, 0, stream>>>(attr, lnW1, lnb1, lnW2, lnb2, lnW3, lnb3,
                                              out + OFF_LANE);

  // 3) ego MLP + GRU + score MLP -> score
  FcArgs fa;
  fa.ego = ego;
  fa.W1 = fcW1; fa.b1 = fcb1; fa.W2 = fcW2; fa.b2 = fcb2; fa.W3 = fcW3; fa.b3 = fcb3;
  fa.Wih = Wih; fa.bih = bih; fa.bhh = bhh; fa.pos = pos;
  fa.sW1 = scW1; fa.sb1 = scb1; fa.sW2 = scW2; fa.sb2 = scb2; fa.sW3 = scW3; fa.sb3 = scb3;
  fa.score_out = out + OFF_SCORE;
  k_fc_gru_score<<<NM_ / 128, 256, 0, stream>>>(fa);

  // 4) pairwise distances/offsets + indices/mask (depends on world from k1)
  k_pairs<<<S_ * M_, 256, 0, stream>>>(world, vmask,
                                       out + OFF_DIST, out + OFF_OFFS,
                                       out + OFF_IDX, out + OFF_MASK);
}
